// MultiHeadAttention_30073361007298
// MI455X (gfx1250) — matrix operand order repaired and checked by reading, not compiled, not run on hardware
//
#include <hip/hip_runtime.h>
#include <hip/hip_bf16.h>
#include <float.h>

// Problem constants (from reference): B=2, S=2048, D=1024, H=16, HD=64
#define B_  2
#define S_  2048
#define D_  1024
#define H_  16
#define HD_ 64
#define BH_ (B_*H_)
#define LROW 2052   // LDS row stride in floats: 2052*4 % 256 == 16 -> no bank aliasing

typedef __bf16 bf16;
typedef __attribute__((ext_vector_type(16))) __bf16 v16bf;
typedef __attribute__((ext_vector_type(8)))  __bf16 v8bf;
typedef __attribute__((ext_vector_type(4)))  __bf16 v4bf;
typedef __attribute__((ext_vector_type(8)))  float  v8f;

static __device__ __forceinline__ bf16 f2bf(float f) { return (bf16)f; }

static __device__ __forceinline__ v8f wmma_bf16(v16bf a, v16bf b, v8f c) {
  return __builtin_amdgcn_wmma_f32_16x16x32_bf16(false, a, false, b,
                                                 (short)0, c, false, false);
}

// A frag (bf16 16x32): lane L row M=L&15; elem j -> k = 16*(L>>4)+j
// B frag (bf16 32x16): lane L col N=L&15; elem j -> k = 16*(L>>4)+j
// C frag (f32 16x16):  acc[r] -> row r+8*(L>>4), col L&15
static __device__ __forceinline__ v16bf load_afrag(const bf16* p, int kk) {
  v8bf lo = *(const v8bf*)(p + kk);
  v8bf hi = *(const v8bf*)(p + kk + 8);
  v16bf a;
#pragma unroll
  for (int j = 0; j < 8; ++j) { a[j] = lo[j]; a[j + 8] = hi[j]; }
  return a;
}

// ---------------------------------------------------------------------------
// Kernel 0: bulk fp32 -> bf16 conversion (memory-bound, one pass).
// ---------------------------------------------------------------------------
__global__ __launch_bounds__(256)
void cvt_bf16_kernel(const float* __restrict__ in, bf16* __restrict__ out, int n4)
{
  const int i = blockIdx.x * 256 + threadIdx.x;
  if (i < n4) {
    const float4 f = ((const float4*)in)[i];
    v4bf o;
    o[0] = f2bf(f.x); o[1] = f2bf(f.y); o[2] = f2bf(f.z); o[3] = f2bf(f.w);
    ((v4bf*)out)[i] = o;
  }
}

// ---------------------------------------------------------------------------
// Shared GEMM core: one wave computes 16x64 (4 accumulators), A frag reused.
// All 4 B fragments get distinct registers so their loads clause together and
// the WMMAs drain the load counter incrementally instead of wait(0) each.
// ---------------------------------------------------------------------------
#define GEMM_CORE(arow, wrow0)                                                \
  v8f acc[4] = {};                                                            \
  for (int kk = 0; kk < D_; kk += 32) {                                       \
    const v16bf a = load_afrag(arow, kk);                                     \
    __builtin_prefetch(arow + kk + 256, 0, 3);                                \
    v16bf bfr[4];                                                             \
_Pragma("unroll")                                                             \
    for (int t = 0; t < 4; ++t)                                               \
      bfr[t] = *(const v16bf*)(wrow0 + ((size_t)(t << 4)) * D_ + kk);         \
_Pragma("unroll")                                                             \
    for (int t = 0; t < 4; ++t) acc[t] = wmma_bf16(a, bfr[t], acc[t]);        \
  }

// ---------------------------------------------------------------------------
// Kernel 1: Y = X @ W^T + b, output head-split as bf16 [BH, S, HD]
//   Block = 8 waves = 2 row-groups x 4 col-groups (32 rows x 256 cols).
// ---------------------------------------------------------------------------
__global__ __launch_bounds__(256)
void proj_qk_kernel(const bf16* __restrict__ Xb, const bf16* __restrict__ Wb,
                    const float* __restrict__ bias, bf16* __restrict__ Yh)
{
  const int lane   = threadIdx.x & 31;
  const int wave   = threadIdx.x >> 5;
  const int laneHi = lane >> 4;
  const int lane16 = lane & 15;
  const int tileM  = (blockIdx.y << 5) + ((wave >> 2) << 4);
  const int tileN  = (blockIdx.x << 8) + ((wave & 3) << 6);
  const int m = tileM + lane16;

  const bf16* arow  = Xb + (size_t)m * D_ + (laneHi << 4);
  const bf16* wrow0 = Wb + (size_t)(tileN + lane16) * D_ + (laneHi << 4);

  GEMM_CORE(arow, wrow0)

#pragma unroll
  for (int t = 0; t < 4; ++t) {
    const int n  = tileN + (t << 4) + lane16;
    const float bvl = bias[n];
    const int h  = n >> 6;   // n / HD
    const int dd = n & 63;   // n % HD
#pragma unroll
    for (int r = 0; r < 8; ++r) {
      const int gm = tileM + r + (laneHi << 3);
      const int b_ = gm >> 11;     // gm / S
      const int s_ = gm & 2047;    // gm % S
      Yh[(((size_t)(b_ * H_ + h) * S_) + s_) * HD_ + dd] = f2bf(acc[t][r] + bvl);
    }
  }
}

// ---------------------------------------------------------------------------
// Kernel 1v: V projection, but output TRANSPOSED: Vt [BH, HD, S] bf16.
//   PV's B fragment wants a V column (fixed d, contiguous k) -> storing V
//   transposed turns phase-3 fragment loads into contiguous b128s.
//   acc[t][r] over r is 8 consecutive sequence positions -> one v8bf store.
// ---------------------------------------------------------------------------
__global__ __launch_bounds__(256)
void proj_v_kernel(const bf16* __restrict__ Xb, const bf16* __restrict__ Wb,
                   const float* __restrict__ bias, bf16* __restrict__ Vt)
{
  const int lane   = threadIdx.x & 31;
  const int wave   = threadIdx.x >> 5;
  const int laneHi = lane >> 4;
  const int lane16 = lane & 15;
  const int tileM  = (blockIdx.y << 5) + ((wave >> 2) << 4);
  const int tileN  = (blockIdx.x << 8) + ((wave & 3) << 6);
  const int m = tileM + lane16;

  const bf16* arow  = Xb + (size_t)m * D_ + (laneHi << 4);
  const bf16* wrow0 = Wb + (size_t)(tileN + lane16) * D_ + (laneHi << 4);

  GEMM_CORE(arow, wrow0)

  const int gm0 = tileM + (laneHi << 3);   // 8 consecutive rows start here
  const int b_  = gm0 >> 11;
  const int s0  = gm0 & 2047;
#pragma unroll
  for (int t = 0; t < 4; ++t) {
    const int n  = tileN + (t << 4) + lane16;
    const float bvl = bias[n];
    const int h  = n >> 6;
    const int dd = n & 63;
    v8bf pk;
#pragma unroll
    for (int r = 0; r < 8; ++r) pk[r] = f2bf(acc[t][r] + bvl);
    *(v8bf*)(Vt + ((size_t)(b_ * H_ + h) * HD_ + dd) * S_ + s0) = pk;
  }
}

// ---------------------------------------------------------------------------
// Kernel 2 (fused): logits -> softmax -> weights write -> P@V, one workgroup
// per (bh, 16-row q-tile). Dynamic LDS holds the full 16 x 2048 logit rows so
// the 537MB weights tensor touches HBM exactly once (the mandatory write).
// ---------------------------------------------------------------------------
__global__ __launch_bounds__(256)
void attn_fused_kernel(const bf16* __restrict__ Qh, const bf16* __restrict__ Kh,
                       const bf16* __restrict__ Vt, float* __restrict__ Wt,
                       bf16* __restrict__ Att)
{
  extern __shared__ float lds[];           // 16 * LROW floats (~128KB)
  const int lane   = threadIdx.x & 31;
  const int wave   = threadIdx.x >> 5;
  const int laneHi = lane >> 4;
  const int lane16 = lane & 15;
  const int bh     = blockIdx.y;
  const int qt     = blockIdx.x;           // q tile index
  const int tileQ  = qt << 4;
  const int gq     = tileQ + lane16;
  const int kLen   = tileQ + 16;           // causal: valid k in [0, kLen)
  const int kLen32 = (kLen + 31) & ~31;

  // ---- Phase 1: logit tiles into LDS (waves stripe over causal k-tiles) ----
  const bf16* qrow = Qh + ((size_t)bh * S_ + gq) * HD_ + (laneHi << 4);
  const v16bf a0 = load_afrag(qrow, 0);
  const v16bf a1 = load_afrag(qrow, 32);

  const float scale = 0.125f;              // 1/sqrt(HD)
  for (int kt = wave; kt <= qt; kt += 8) {
    const int gk = (kt << 4) + lane16;
    const bf16* krow = Kh + ((size_t)bh * S_ + gk) * HD_ + (laneHi << 4);
    v16bf b0 = *(const v16bf*)(krow);
    v16bf b1 = *(const v16bf*)(krow + 32);
    v8f acc = {};
    acc = wmma_bf16(a0, b0, acc);
    acc = wmma_bf16(a1, b1, acc);
#pragma unroll
    for (int r = 0; r < 8; ++r) {
      const int row = r + (laneHi << 3);
      float vv = acc[r] * scale;
      if (gk > tileQ + row) vv = -FLT_MAX; // causal mask (diagonal tile)
      lds[row * LROW + gk] = vv;
    }
  }
  __syncthreads();

  // ---- Phase 2: softmax per row in LDS; stream weights (and zeros) out ----
  float* wout = Wt + (size_t)bh * S_ * S_;
  for (int row = wave; row < 16; row += 8) {
    float* lrow = lds + row * LROW;
    float m = -FLT_MAX;
    for (int c = lane; c < kLen; c += 32) m = fmaxf(m, lrow[c]);
#pragma unroll
    for (int off = 16; off; off >>= 1) m = fmaxf(m, __shfl_xor(m, off, 32));
    float s = 0.f;
    for (int c = lane; c < kLen; c += 32) {
      float e = __expf(lrow[c] - m);
      lrow[c] = e; s += e;
    }
#pragma unroll
    for (int off = 16; off; off >>= 1) s += __shfl_xor(s, off, 32);
    const float inv = 1.0f / s;
    float* grow = wout + (size_t)(tileQ + row) * S_;
    for (int c = lane; c < kLen; c += 32) {
      float wv = lrow[c] * inv;
      lrow[c] = wv;                        // final P for phase 3
      grow[c] = wv;                        // weights output
    }
    for (int c = kLen + lane; c < kLen32; c += 32) lrow[c] = 0.f; // pad for PV
    for (int c = kLen + lane; c < S_; c += 32) grow[c] = 0.f;     // upper tri
  }
  __syncthreads();

  // ---- Phase 3: att = P @ V, split-K across wave pairs ----
  // wave w: d-tile = w&3 (HD=64 -> 4 tiles), k-half = w>>2.
  const int mid = ((kLen32 >> 1) + 31) & ~31;
  const int dt  = (wave & 3) << 4;
  const int k0  = (wave >> 2) ? mid : 0;
  const int k1  = (wave >> 2) ? kLen32 : mid;
  const int gd  = dt + lane16;
  // Vt [BH, HD, S]: B frag (k,n=d) = Vt[d][k] -> contiguous along k per lane
  const bf16* vtrow = Vt + ((size_t)bh * HD_ + gd) * S_ + (laneHi << 4);
  const float* arow = lds + lane16 * LROW + (laneHi << 4);

  v8f acc = {};
  for (int kk = k0; kk < k1; kk += 32) {
    v16bf a;
#pragma unroll
    for (int j = 0; j < 8; ++j) {
      a[j]     = f2bf(arow[kk + j]);
      a[j + 8] = f2bf(arow[kk + 8 + j]);
    }
    const v16bf b = *(const v16bf*)(vtrow + kk);
    acc = wmma_bf16(a, b, acc);
  }
  __syncthreads();                         // done reading logits region

  float* scratch = lds;                    // reuse LDS for partial combine
  if (wave >= 4) {
#pragma unroll
    for (int r = 0; r < 8; ++r)
      scratch[(((wave & 3) << 5) + lane) * 8 + r] = acc[r];
  }
  __syncthreads();
  if (wave < 4) {
#pragma unroll
    for (int r = 0; r < 8; ++r)
      acc[r] += scratch[(((wave & 3) << 5) + lane) * 8 + r];
#pragma unroll
    for (int r = 0; r < 8; ++r) {
      const int row = tileQ + r + (laneHi << 3);
      Att[((size_t)bh * S_ + row) * HD_ + gd] = f2bf(acc[r]);
    }
  }
}

// ---------------------------------------------------------------------------
// Kernel 3: out = merge_heads(att) @ Wo^T + bo -> fp32 [B, S, D]
// ---------------------------------------------------------------------------
__global__ __launch_bounds__(256)
void out_proj_kernel(const bf16* __restrict__ Att, const bf16* __restrict__ Wob,
                     const float* __restrict__ bo, float* __restrict__ Out)
{
  const int lane   = threadIdx.x & 31;
  const int wave   = threadIdx.x >> 5;
  const int laneHi = lane >> 4;
  const int lane16 = lane & 15;
  const int tileM  = (blockIdx.y << 5) + ((wave >> 2) << 4);
  const int tileN  = (blockIdx.x << 8) + ((wave & 3) << 6);
  const int m = tileM + lane16;
  const int b_ = m >> 11, s_ = m & 2047;

  const bf16* wrow0 = Wob + (size_t)(tileN + lane16) * D_ + (laneHi << 4);
  const bf16* abase = Att + ((size_t)b_ * H_ * S_ + s_) * HD_ + (laneHi << 4);

  v8f acc[4] = {};
  for (int kk = 0; kk < D_; kk += 32) {
    const int h = kk >> 6;                        // head const over 32-chunk
    const v16bf a = load_afrag(abase + ((size_t)h * S_) * HD_, kk & 63);
    v16bf bfr[4];
#pragma unroll
    for (int t = 0; t < 4; ++t)
      bfr[t] = *(const v16bf*)(wrow0 + ((size_t)(t << 4)) * D_ + kk);
#pragma unroll
    for (int t = 0; t < 4; ++t) acc[t] = wmma_bf16(a, bfr[t], acc[t]);
  }

#pragma unroll
  for (int t = 0; t < 4; ++t) {
    const int n   = tileN + (t << 4) + lane16;
    const float bvl = bo[n];
#pragma unroll
    for (int r = 0; r < 8; ++r) {
      const int gm = tileM + r + (laneHi << 3);
      Out[(size_t)gm * D_ + n] = acc[t][r] + bvl;
    }
  }
}

// ---------------------------------------------------------------------------
extern "C" void kernel_launch(void* const* d_in, const int* in_sizes, int n_in,
                              void* d_out, int out_size, void* d_ws, size_t ws_size,
                              hipStream_t stream) {
  const float* q  = (const float*)d_in[0];
  const float* k  = (const float*)d_in[1];
  const float* v  = (const float*)d_in[2];
  // d_in[3] = causal mask (bool) — structure known, not needed
  const float* Wq = (const float*)d_in[4];
  const float* bq = (const float*)d_in[5];
  const float* Wk = (const float*)d_in[6];
  const float* bk = (const float*)d_in[7];
  const float* Wv = (const float*)d_in[8];
  const float* bv = (const float*)d_in[9];
  const float* Wo = (const float*)d_in[10];
  const float* bo = (const float*)d_in[11];

  float* out     = (float*)d_out;                       // [B,S,D]
  float* weights = out + (size_t)B_ * S_ * D_;          // [BH,S,S]

  const size_t xElems = (size_t)B_ * S_ * D_;           // 4M
  const size_t wElems = (size_t)D_ * D_;                // 1M
  const size_t hElems = (size_t)BH_ * S_ * HD_;         // 4M

  bf16* p = (bf16*)d_ws;
  bf16* xq  = p; p += xElems;
  bf16* xk  = p; p += xElems;
  bf16* xv  = p; p += xElems;
  bf16* wqb = p; p += wElems;
  bf16* wkb = p; p += wElems;
  bf16* wvb = p; p += wElems;
  bf16* wob = p; p += wElems;
  bf16* qh  = p; p += hElems;
  bf16* kh  = p; p += hElems;
  bf16* vt  = p; p += hElems;                           // transposed V
  bf16* att = p;                                        // total 64MB of ws

  dim3 blk(256);

  // Pre-convert all GEMM operands once (memory-bound)
  const int x4 = (int)(xElems / 4), w4 = (int)(wElems / 4);
  cvt_bf16_kernel<<<dim3((x4 + 255) / 256), blk, 0, stream>>>(q, xq, x4);
  cvt_bf16_kernel<<<dim3((x4 + 255) / 256), blk, 0, stream>>>(k, xk, x4);
  cvt_bf16_kernel<<<dim3((x4 + 255) / 256), blk, 0, stream>>>(v, xv, x4);
  cvt_bf16_kernel<<<dim3((w4 + 255) / 256), blk, 0, stream>>>(Wq, wqb, w4);
  cvt_bf16_kernel<<<dim3((w4 + 255) / 256), blk, 0, stream>>>(Wk, wkb, w4);
  cvt_bf16_kernel<<<dim3((w4 + 255) / 256), blk, 0, stream>>>(Wv, wvb, w4);
  cvt_bf16_kernel<<<dim3((w4 + 255) / 256), blk, 0, stream>>>(Wo, wob, w4);

  dim3 gProj(D_ / 256, (B_ * S_) / 32);                 // (4, 128)
  proj_qk_kernel<<<gProj, blk, 0, stream>>>(xq, wqb, bq, qh);
  proj_qk_kernel<<<gProj, blk, 0, stream>>>(xk, wkb, bk, kh);
  proj_v_kernel <<<gProj, blk, 0, stream>>>(xv, wvb, bv, vt);

  dim3 gAtt(S_ / 16, BH_);                              // (128, 32)
  const size_t ldsBytes = (size_t)16 * LROW * sizeof(float);  // ~128KB
  attn_fused_kernel<<<gAtt, blk, ldsBytes, stream>>>(qh, kh, vt, weights, att);

  out_proj_kernel<<<gProj, blk, 0, stream>>>(att, wob, bo, out);
}
